// PoolHiddenNet_6949257085630
// MI455X (gfx1250) — compile-verified, matrix operand order checked by reading, not run on hardware
//
#include <hip/hip_runtime.h>
#include <hip/hip_bf16.h>
#include <math.h>

// ---------------------------------------------------------------------------
// PoolHiddenNet fused pipeline for gfx1250 (MI455X), wave32 + WMMA.
//
// S=64 scenes, P=32 peds, H=E=64, D1=512, D2=1024; per scene M=P*P=1024 rows.
// Global GEMMs: [65536x128]@[128x512] and [65536x512]@[512x1024] (~77 GFLOP).
//
// Exact simplifications:
//  * b1/b2 dropped: per-channel constants are removed by the following BN.
//  * max_j relu(z) == relu(max_j z); BN affine applied per element pre-max.
// Precision: GEMM operands f16, accumulation f32 (WMMA), BN stats f32.
//
// GEMM tiling: 256 threads = 8 waves; each wave owns a 32(M)x64(N) tile.
// Per K-chunk of 32: both A fragments and ALL FOUR B fragments are loaded
// up front (batched B128 load clauses), then 8 WMMAs run while later loads
// are still in flight -> incremental s_wait_loadcnt instead of full drains.
// Per-lane register layouts per CDNA5 ISA 7.12.2.
// ---------------------------------------------------------------------------

typedef __attribute__((ext_vector_type(16))) _Float16 v16h;
typedef __attribute__((ext_vector_type(8)))  _Float16 v8h;
typedef __attribute__((ext_vector_type(8)))  float    v8f;

#define S_  64
#define P_  32
#define MROWS 65536   // S_*P_*P_
#define KX  128       // E + H
#define D1_ 512
#define D2_ 1024
#define EPS_ 1e-5f

union AV { v16h v; unsigned int u[8]; };

// ---------------------------------------------------------------------------
// 0) weight transpose + f32->f16:  Wt[n*K + k] = W[k*N + n]
// ---------------------------------------------------------------------------
__global__ void k_transpose_f16(const float* __restrict__ W,
                                _Float16* __restrict__ Wt, int K, int N) {
  int i = blockIdx.x * blockDim.x + threadIdx.x;      // indexes Wt, size K*N
  if (i >= K * N) return;
  int k = i % K;
  int n = i / K;
  Wt[i] = (_Float16)W[(size_t)k * N + n];
}

// ---------------------------------------------------------------------------
// 1) build X f16 [MROWS x 128]: row (s,k,j) -> [emb(pos_j - pos_k) | h_j]
// ---------------------------------------------------------------------------
__global__ void k_build_x(const float* __restrict__ h,      // [N, 64]
                          const float* __restrict__ pos,    // [N, 2]
                          const float* __restrict__ We,     // [2, 64]
                          const float* __restrict__ be,     // [64]
                          _Float16* __restrict__ X) {
  size_t i = (size_t)blockIdx.x * blockDim.x + threadIdx.x; // over MROWS*128
  int c  = (int)(i & 127);
  size_t row = i >> 7;
  int j  = (int)(row & 31);
  int kk = (int)((row >> 5) & 31);
  int s  = (int)(row >> 10);
  float v;
  if (c < 64) {
    float dx = pos[(s * 32 + j) * 2 + 0] - pos[(s * 32 + kk) * 2 + 0];
    float dy = pos[(s * 32 + j) * 2 + 1] - pos[(s * 32 + kk) * 2 + 1];
    v = fmaf(dx, We[c], fmaf(dy, We[64 + c], be[c]));
  } else {
    v = h[(size_t)(s * 32 + j) * 64 + (c - 64)];
  }
  X[i] = (_Float16)v;
}

// ---------------------------------------------------------------------------
// 2) WMMA GEMM:  C[M x N] = A[M x K] @ Bt[N x K]^T   (f16 in, f32 accum)
//    Wave tile 32(M) x 64(N): acc[2][4] v8f; B fragments reused for both
//    M-subtiles. Block covers 256 rows x 64 cols.
// ---------------------------------------------------------------------------
__global__ __launch_bounds__(256) void k_gemm_wmma(
    const _Float16* __restrict__ A, const _Float16* __restrict__ Bt,
    _Float16* __restrict__ C, int K, int N) {
  const int wave = threadIdx.x >> 5;
  const int lane = threadIdx.x & 31;
  const int mlo  = lane & 15;        // row (A) / col (B,D) within tile
  const int hi   = lane >> 4;        // K half-select for A/B, M half for D
  const int row0 = blockIdx.x * 256 + wave * 32;
  const int col0 = blockIdx.y * 64;

  const v8f vzero = {0.f, 0.f, 0.f, 0.f, 0.f, 0.f, 0.f, 0.f};
  v8f acc[2][4];
#pragma unroll
  for (int mt = 0; mt < 2; ++mt)
#pragma unroll
    for (int t = 0; t < 4; ++t) acc[mt][t] = vzero;

  for (int kc = 0; kc < K; kc += 32) {
    // A fragments: lane holds row (row0 + 16*mt + mlo), K pairs {kb,kb+1}
    AV a[2];
#pragma unroll
    for (int mt = 0; mt < 2; ++mt) {
      const _Float16* arow = A + (size_t)(row0 + mt * 16 + mlo) * K + kc + hi * 8;
#pragma unroll
      for (int p = 0; p < 8; ++p) {
        int kb = (p < 4) ? (p * 2) : (16 + (p - 4) * 2);
        a[mt].u[p] = *(const unsigned int*)(arow + kb);
      }
    }
    // ALL B fragments of this K-chunk up front (batched load clauses)
    AV b[4];
#pragma unroll
    for (int t = 0; t < 4; ++t) {
      const _Float16* bcol = Bt + (size_t)(col0 + t * 16 + mlo) * K + kc + hi * 8;
#pragma unroll
      for (int p = 0; p < 8; ++p) {
        int kb = (p < 4) ? (p * 2) : (16 + (p - 4) * 2);
        b[t].u[p] = *(const unsigned int*)(bcol + kb);
      }
    }
    // 8 WMMAs; compiler can wait incrementally as fragments land
#pragma unroll
    for (int t = 0; t < 4; ++t) {
      acc[0][t] = __builtin_amdgcn_wmma_f32_16x16x32_f16(
          false, a[0].v, false, b[t].v, (short)0, acc[0][t], false, false);
      acc[1][t] = __builtin_amdgcn_wmma_f32_16x16x32_f16(
          false, a[1].v, false, b[t].v, (short)0, acc[1][t], false, false);
    }
  }

  // D layout: lane -> col (col0+16t+mlo); vgpr v -> row (v + 8*hi)
#pragma unroll
  for (int mt = 0; mt < 2; ++mt) {
#pragma unroll
    for (int t = 0; t < 4; ++t) {
      int n = col0 + t * 16 + mlo;
#pragma unroll
      for (int v = 0; v < 8; ++v) {
        int mm = mt * 16 + v + 8 * hi;
        C[(size_t)(row0 + mm) * N + n] = (_Float16)acc[mt][t][v];
      }
    }
  }
}

// ---------------------------------------------------------------------------
// 3) per-(scene,channel) BN stats over 1024 rows -> scale a, shift b
// ---------------------------------------------------------------------------
__global__ void k_bn_stats(const _Float16* __restrict__ Y,
                           const float* __restrict__ g,
                           const float* __restrict__ be,
                           float* __restrict__ Aout, float* __restrict__ Bout,
                           int Nc) {
  int idx = blockIdx.x * blockDim.x + threadIdx.x;  // s*Nc + c
  if (idx >= S_ * Nc) return;
  int s = idx / Nc, c = idx % Nc;
  const _Float16* p = Y + (size_t)s * 1024 * Nc + c;
  float sum = 0.f, sq = 0.f;
  for (int r = 0; r < 1024; ++r) {
    float v = (float)p[(size_t)r * Nc];
    sum += v;
    sq = fmaf(v, v, sq);
  }
  float mu  = sum * (1.f / 1024.f);
  float var = sq * (1.f / 1024.f) - mu * mu;
  float a   = g[c] * rsqrtf(var + EPS_);
  Aout[idx] = a;
  Bout[idx] = be[c] - mu * a;
}

// ---------------------------------------------------------------------------
// 4) in-place BN + relu, vectorized 8 f16 per thread (f32 math)
// ---------------------------------------------------------------------------
__global__ void k_bn_relu(_Float16* __restrict__ Y,
                          const float* __restrict__ Aa,
                          const float* __restrict__ Bb, int Nc) {
  size_t i = (size_t)blockIdx.x * blockDim.x + threadIdx.x;  // 8-elt groups
  if (i >= (size_t)MROWS * Nc / 8) return;
  size_t e0 = i * 8;
  int c0 = (int)(e0 % Nc);
  int s  = (int)((e0 / Nc) >> 10);
  const float* ap = Aa + (size_t)s * Nc + c0;
  const float* bp = Bb + (size_t)s * Nc + c0;
  v8h y = *(const v8h*)(Y + e0);
  v8h r;
#pragma unroll
  for (int e = 0; e < 8; ++e) {
    float v = fmaf((float)y[e], ap[e], bp[e]);
    r[e] = (_Float16)(v > 0.f ? v : 0.f);
  }
  *(v8h*)(Y + e0) = r;
}

// ---------------------------------------------------------------------------
// 5) fused BN + max over j + relu -> out f32 [2048 x 1024]; 8 ch per thread
// ---------------------------------------------------------------------------
__global__ void k_pool(const _Float16* __restrict__ Y2,
                       const float* __restrict__ Aa,
                       const float* __restrict__ Bb,
                       float* __restrict__ out) {
  int idx = blockIdx.x * blockDim.x + threadIdx.x;  // over S_*P_*(D2_/8)
  if (idx >= S_ * P_ * (D2_ / 8)) return;
  int c8 = idx & (D2_ / 8 - 1);     // 0..127
  int sk = idx >> 7;                // s*32 + k
  int s  = sk >> 5;
  int c0 = c8 * 8;
  float a[8], b[8], mx[8];
#pragma unroll
  for (int e = 0; e < 8; ++e) {
    a[e]  = Aa[(size_t)s * D2_ + c0 + e];
    b[e]  = Bb[(size_t)s * D2_ + c0 + e];
    mx[e] = -INFINITY;
  }
  const _Float16* p = Y2 + (size_t)sk * 32 * D2_ + c0;
  for (int j = 0; j < 32; ++j) {
    v8h y = *(const v8h*)(p + (size_t)j * D2_);
#pragma unroll
    for (int e = 0; e < 8; ++e)
      mx[e] = fmaxf(mx[e], fmaf((float)y[e], a[e], b[e]));
  }
  v8f o;
#pragma unroll
  for (int e = 0; e < 8; ++e) o[e] = fmaxf(mx[e], 0.f);  // relu(max)==max(relu)
  *(v8f*)(out + (size_t)sk * D2_ + c0) = o;
}

// ---------------------------------------------------------------------------
// host launcher
// ---------------------------------------------------------------------------
extern "C" void kernel_launch(void* const* d_in, const int* in_sizes, int n_in,
                              void* d_out, int out_size, void* d_ws, size_t ws_size,
                              hipStream_t stream) {
  // inputs: 0 h_states, 1 seq_start_end, 2 end_pos, 3 rel_pos,
  //         4 W_embed, 5 b_embed, 6 W1, 7 b1, 8 g1, 9 be1,
  //         10 W2, 11 b2, 12 g2, 13 be2
  const float* h_states = (const float*)d_in[0];
  const float* end_pos  = (const float*)d_in[2];
  const float* W_embed  = (const float*)d_in[4];
  const float* b_embed  = (const float*)d_in[5];
  const float* W1 = (const float*)d_in[6];
  const float* g1 = (const float*)d_in[8];
  const float* be1 = (const float*)d_in[9];
  const float* W2 = (const float*)d_in[10];
  const float* g2 = (const float*)d_in[12];
  const float* be2 = (const float*)d_in[13];
  float* out = (float*)d_out;

  // ---- workspace carve-up (all offsets 256B aligned) ----
  char* ws = (char*)d_ws;
  size_t off = 0;
  _Float16* W1t = (_Float16*)(ws + off); off += (size_t)KX  * D1_ * 2;   // 128 KB
  _Float16* W2t = (_Float16*)(ws + off); off += (size_t)D1_ * D2_ * 2;   // 1 MB
  _Float16* Xh  = (_Float16*)(ws + off); off += (size_t)MROWS * KX  * 2; // 16 MB
  _Float16* Y1h = (_Float16*)(ws + off); off += (size_t)MROWS * D1_ * 2; // 64 MB (reused as X1)
  _Float16* Y2h = (_Float16*)(ws + off); off += (size_t)MROWS * D2_ * 2; // 128 MB
  float* A1 = (float*)(ws + off); off += (size_t)S_ * D1_ * 4;
  float* B1 = (float*)(ws + off); off += (size_t)S_ * D1_ * 4;
  float* A2 = (float*)(ws + off); off += (size_t)S_ * D2_ * 4;
  float* B2 = (float*)(ws + off); off += (size_t)S_ * D2_ * 4;
  (void)ws_size; (void)in_sizes; (void)n_in; (void)out_size;

  const int T = 256;

  // 0) weights -> transposed f16
  k_transpose_f16<<<(KX * D1_ + T - 1) / T, T, 0, stream>>>(W1, W1t, KX, D1_);
  k_transpose_f16<<<(D1_ * D2_ + T - 1) / T, T, 0, stream>>>(W2, W2t, D1_, D2_);

  // 1) build X
  {
    size_t n = (size_t)MROWS * KX;
    k_build_x<<<(unsigned)((n + T - 1) / T), T, 0, stream>>>(h_states, end_pos,
                                                             W_embed, b_embed, Xh);
  }

  // 2) GEMM1: Y1 = X @ W1   (bias absorbed by BN)
  {
    dim3 grid(MROWS / 256, D1_ / 64);
    k_gemm_wmma<<<grid, T, 0, stream>>>(Xh, W1t, Y1h, KX, D1_);
  }

  // 3) BN1 stats, 4) BN1+relu in place (Y1h becomes X1)
  k_bn_stats<<<(S_ * D1_ + T - 1) / T, T, 0, stream>>>(Y1h, g1, be1, A1, B1, D1_);
  {
    size_t n = (size_t)MROWS * D1_ / 8;
    k_bn_relu<<<(unsigned)((n + T - 1) / T), T, 0, stream>>>(Y1h, A1, B1, D1_);
  }

  // 5) GEMM2: Y2 = X1 @ W2
  {
    dim3 grid(MROWS / 256, D2_ / 64);
    k_gemm_wmma<<<grid, T, 0, stream>>>(Y1h, W2t, Y2h, D1_, D2_);
  }

  // 6) BN2 stats, 7) fused BN2 + max-pool(j) + relu -> out
  k_bn_stats<<<(S_ * D2_ + T - 1) / T, T, 0, stream>>>(Y2h, g2, be2, A2, B2, D2_);
  k_pool<<<(S_ * P_ * (D2_ / 8) + T - 1) / T, T, 0, stream>>>(Y2h, A2, B2, out);
}
